// NodeUpdate_29661044146683
// MI455X (gfx1250) — compile-verified, matrix operand order checked by reading, not run on hardware
//
#include <hip/hip_runtime.h>
#include <cstdint>
#include <cstddef>

// ---------------- problem constants ----------------
#define BATCH 16
#define AT    96
#define NBR   12
#define FN    64
#define MROWS 18432      // BATCH*AT*NBR
#define NBA   1536       // BATCH*AT
#define I1    128        // Fn+Fe
#define N1    512        // H1
#define I2    512
#define N2    128        // 2*Fn
#define K1    (I1*8)     // 1024
#define K2    (I2*8)     // 4096

// ---------------- workspace layout (bytes) ----------------
static constexpr size_t O_SCAL = 0;                               // 4 encoded uints (min1,max1,min2,max2)
static constexpr size_t O_W1   = 256;                             // f16 [512][1024]  = 1 MiB
static constexpr size_t O_W2   = O_W1 + (size_t)N1*K1*2;          // f16 [128][4096]  = 1 MiB
static constexpr size_t O_RS1  = O_W2 + (size_t)N2*K2*2;          // f32 [512]
static constexpr size_t O_RS2  = O_RS1 + 2048;                    // f32 [128]
static constexpr size_t O_Q1   = O_RS2 + 512;                     // u8  [M][128]
static constexpr size_t O_Q2   = O_Q1 + (size_t)MROWS*I1;         // u8  [M][512]
static constexpr size_t O_H    = O_Q2 + (size_t)MROWS*I2;         // f32 [M][512]
static constexpr size_t O_C2   = O_H  + (size_t)MROWS*I2*4;       // f32 [M][128]
static constexpr size_t O_NS   = O_C2 + (size_t)MROWS*N2*4;       // f32 [1536][64]

// ---------------- WMMA types ----------------
typedef __attribute__((ext_vector_type(16))) _Float16 v16h;
typedef __attribute__((ext_vector_type(8)))  _Float16 v8h;
typedef __attribute__((ext_vector_type(8)))  float    v8f;

union Frag16 { v16h v; v8h h[2]; };

__device__ __forceinline__ v8f wmma_f16(const Frag16& a, const Frag16& b, v8f c) {
  return __builtin_amdgcn_wmma_f32_16x16x32_f16(false, a.v, false, b.v, (short)0, c, false, false);
}

// CDNA5 async global->LDS copy (ASYNCcnt-tracked); LDS dest offset in a VGPR,
// 64-bit global address in a VGPR pair (GV addressing mode).
__device__ __forceinline__ void async_b128(unsigned lds_off, const void* gaddr) {
  asm volatile("global_load_async_to_lds_b128 %0, %1, off"
               :: "v"(lds_off), "v"((unsigned long long)(uintptr_t)gaddr)
               : "memory");
}
__device__ __forceinline__ void wait_async0() {
  asm volatile("s_wait_asynccnt 0x0" ::: "memory");
}

// ordered-uint encoding: enc is monotonic in float value -> integer atomicMin/Max works
__device__ __forceinline__ unsigned encf(float f) {
  unsigned u = __float_as_uint(f);
  return (u & 0x80000000u) ? ~u : (u | 0x80000000u);
}
__device__ __forceinline__ float decf(unsigned e) {
  unsigned u = (e & 0x80000000u) ? (e ^ 0x80000000u) : ~e;
  return __uint_as_float(u);
}

// ---------------- small kernels ----------------
__global__ void init_kernel(unsigned* scal) {
  if (threadIdx.x == 0) { scal[0] = 0xFFFFFFFFu; scal[1] = 0u; scal[2] = 0xFFFFFFFFu; scal[3] = 0u; }
}

__global__ void minmax_kernel(const float* __restrict__ a, int na,
                              const float* __restrict__ b, int nb,
                              unsigned* __restrict__ scal) {
  __shared__ float smin[256], smax[256];
  const int total = na + nb;
  float lmin = 3.4e38f, lmax = -3.4e38f;
  for (int i = blockIdx.x * blockDim.x + threadIdx.x; i < total; i += gridDim.x * blockDim.x) {
    float v = (i < na) ? a[i] : b[i - na];
    lmin = fminf(lmin, v); lmax = fmaxf(lmax, v);
  }
  smin[threadIdx.x] = lmin; smax[threadIdx.x] = lmax; __syncthreads();
  for (int s = 128; s > 0; s >>= 1) {
    if ((int)threadIdx.x < s) {
      smin[threadIdx.x] = fminf(smin[threadIdx.x], smin[threadIdx.x + s]);
      smax[threadIdx.x] = fmaxf(smax[threadIdx.x], smax[threadIdx.x + s]);
    }
    __syncthreads();
  }
  if (threadIdx.x == 0) { atomicMin(&scal[0], encf(smin[0])); atomicMax(&scal[1], encf(smax[0])); }
}

// W'[o][p*I+i] = 2^p * (w[o,i] + 0.1*eps[p,o,i]*|w[o,i]|), stored f16 row-major [O][K]
__global__ void prepw_kernel(const float* __restrict__ w, const float* __restrict__ eps,
                             _Float16* __restrict__ wt, int O, int I) {
  const int K = I * 8;
  long id = (long)blockIdx.x * blockDim.x + threadIdx.x;
  if (id >= (long)O * K) return;
  int o = (int)(id / K), k = (int)(id % K);
  int p = k / I, i = k % I;
  float wv = w[(size_t)o * I + i];
  float nv = wv + 0.1f * eps[(size_t)p * O * I + (size_t)o * I + i] * fabsf(wv);
  wt[(size_t)o * K + k] = (_Float16)(nv * (float)(1 << p));
}

__global__ void rowsum_kernel(const float* __restrict__ src, const float* __restrict__ epsl,
                              float* __restrict__ rs, int O, int I, int noisy) {
  int o = blockIdx.x * blockDim.x + threadIdx.x;
  if (o >= O) return;
  float s = 0.0f;
  for (int i = 0; i < I; ++i) {
    float w = src[(size_t)o * I + i];
    if (noisy) w += 0.1f * epsl[(size_t)o * I + i] * fabsf(w);
    s += w;
  }
  rs[o] = s;
}

__global__ void quant1_kernel(const float* __restrict__ node, const float* __restrict__ edge,
                              unsigned char* __restrict__ q1, const unsigned* __restrict__ scal) {
  int id = blockIdx.x * blockDim.x + threadIdx.x;
  if (id >= MROWS * I1) return;
  int m = id / I1, i = id % I1;
  int ba = m / NBR;
  float x = (i < FN) ? node[(size_t)ba * FN + i] : edge[(size_t)m * FN + (i - FN)];
  float mn = decf(scal[0]), mx = decf(scal[1]);
  float f = (x - mn) / (mx - mn) * 255.0f;
  f = fminf(fmaxf(f, 0.0f), 255.0f);
  q1[id] = (unsigned char)f;   // truncating cast, as in reference
}

__global__ void quant2_kernel(const float* __restrict__ h, unsigned char* __restrict__ q2,
                              const unsigned* __restrict__ scal) {
  size_t id = (size_t)blockIdx.x * blockDim.x + threadIdx.x;
  if (id >= (size_t)MROWS * I2) return;
  float mn = decf(scal[2]), mx = decf(scal[3]);
  float f = (h[id] - mn) / (mx - mn) * 255.0f;
  f = fminf(fmaxf(f, 0.0f), 255.0f);
  q2[id] = (unsigned char)f;
}

// ---------------- WMMA GEMM ----------------
// A[m, p*I+i] = bit_p(q[m,i]) in {0,1}; B = wt [N][K] f16 (plane scale folded).
// Block tile 64(M) x 128(N), Kc=64, double-buffered LDS. 256 threads = 8 wave32s
// laid out 2(M) x 4(N); each wave owns a 32x32 tile = 4 f32 accumulators.
// B tiles are staged with global_load_async_to_lds_b128 (ASYNCcnt), A tiles are
// bit-expanded in-register and ds-stored; both overlap the WMMA compute of the
// previous stage.
template <int IFEAT, int NTOT, bool G1>
__global__ __launch_bounds__(256) void gemm_kernel(const unsigned char* __restrict__ q,
                                                   const _Float16* __restrict__ wt,
                                                   const float* __restrict__ rs,
                                                   float* __restrict__ outbuf,
                                                   unsigned* __restrict__ scal) {
  constexpr int K      = IFEAT * 8;
  constexpr int NSTAGE = K / 64;
  constexpr int LDA = 72;   // padded stride (halfs) -> conflict-free ds_load_b128
  constexpr int LDB = 72;
  __shared__ __align__(16) _Float16 As[2][64 * LDA];
  __shared__ __align__(16) _Float16 Bs[2][128 * LDB];
  __shared__ float smin[256], smax[256];

  const int m0   = blockIdx.x * 64;
  const int n0   = blockIdx.y * 128;
  const int tid  = threadIdx.x;
  const int lane = tid & 31;
  const int wave = tid >> 5;
  const int mw   = wave >> 2;   // 0..1
  const int nw   = wave & 3;    // 0..3
  const int l16  = lane & 15;
  const int hh   = lane >> 4;

  float mn, mx;
  if (G1) { mn = decf(scal[0]); mx = decf(scal[1]); }
  else    { mn = decf(scal[2]); mx = decf(scal[3]); }
  const float scale = (mx - mn) * (1.0f / 255.0f);

  v8f acc00 = {}, acc01 = {}, acc10 = {}, acc11 = {};

  const int ar = tid >> 2;          // A staging row 0..63
  const int ac = (tid & 3) * 16;    // A staging col (16 bytes)
  const int br = tid >> 3;          // B staging row base 0..31 (4 chunks of 32 rows)
  const int bseg = tid & 7;         // 16B segment within 128B row

  auto stageA = [&](int s, int buf) {
    const int k0 = s * 64;
    const int p  = k0 / IFEAT;
    const int ib = k0 % IFEAT;
    uint4 qb = *(const uint4*)(q + (size_t)(m0 + ar) * IFEAT + ib + ac);
    const unsigned char* qc = (const unsigned char*)&qb;
    unsigned* dst = (unsigned*)&As[buf][ar * LDA + ac];
#pragma unroll
    for (int j = 0; j < 8; ++j) {   // pack two f16 lanes per dword: bit -> 1.0h
      unsigned b0 = (qc[2 * j]     >> p) & 1;
      unsigned b1 = (qc[2 * j + 1] >> p) & 1;
      dst[j] = (b0 ? 0x00003C00u : 0u) | (b1 ? 0x3C000000u : 0u);
    }
  };
  auto stageB = [&](int s, int buf) {
    const int k0 = s * 64;
#pragma unroll
    for (int c = 0; c < 4; ++c) {
      const int row = c * 32 + br;
      const void* g = wt + (size_t)(n0 + row) * K + k0 + bseg * 8;
      unsigned l = (unsigned)(uintptr_t)&Bs[buf][row * LDB + bseg * 8];
      async_b128(l, g);
    }
  };
  auto compute = [&](int buf) {
#pragma unroll
    for (int ks = 0; ks < 2; ++ks) {
      const int kb = ks * 32;
      Frag16 a0, a1, b0, b1;
      const _Float16* ap0 = &As[buf][(mw * 32 + l16) * LDA + kb + hh * 8];
      a0.h[0] = *(const v8h*)ap0;  a0.h[1] = *(const v8h*)(ap0 + 16);
      const _Float16* ap1 = ap0 + 16 * LDA;
      a1.h[0] = *(const v8h*)ap1;  a1.h[1] = *(const v8h*)(ap1 + 16);
      const _Float16* bp0 = &Bs[buf][(nw * 32 + l16) * LDB + kb + hh * 8];
      b0.h[0] = *(const v8h*)bp0;  b0.h[1] = *(const v8h*)(bp0 + 16);
      const _Float16* bp1 = bp0 + 16 * LDB;
      b1.h[0] = *(const v8h*)bp1;  b1.h[1] = *(const v8h*)(bp1 + 16);
      acc00 = wmma_f16(a0, b0, acc00);
      acc01 = wmma_f16(a0, b1, acc01);
      acc10 = wmma_f16(a1, b0, acc10);
      acc11 = wmma_f16(a1, b1, acc11);
    }
  };

  stageA(0, 0);
  stageB(0, 0);
  wait_async0();
  __syncthreads();
  for (int s = 0; s < NSTAGE; ++s) {
    const int cur = s & 1;
    if (s + 1 < NSTAGE) {
      stageA(s + 1, cur ^ 1);
      stageB(s + 1, cur ^ 1);
      if (s + 2 < NSTAGE) {   // global_prefetch_b8 of next-next A bytes
        __builtin_prefetch(q + (size_t)(m0 + ar) * IFEAT + (((s + 2) * 64) % IFEAT) + ac, 0, 0);
      }
    }
    compute(cur);
    wait_async0();    // async B copies for stage s+1 landed in LDS
    __syncthreads();  // + everyone done reading buffer `cur`
  }

  // ---- epilogue: C/D layout -> row = v + 8*hh, col = l16
  const int rowb = m0 + mw * 32 + hh * 8;
  const int cb   = n0 + nw * 32 + l16;
  const float bias0 = mn * rs[cb];
  const float bias1 = mn * rs[cb + 16];
  if (G1) {
    float lmin = 3.4e38f, lmax = -3.4e38f;
#pragma unroll
    for (int mi = 0; mi < 2; ++mi) {
#pragma unroll
      for (int ni = 0; ni < 2; ++ni) {
        v8f a = (mi == 0) ? ((ni == 0) ? acc00 : acc01) : ((ni == 0) ? acc10 : acc11);
        const float bias = (ni == 0) ? bias0 : bias1;
        const int col = cb + ni * 16;
#pragma unroll
        for (int v = 0; v < 8; ++v) {
          const size_t row = (size_t)(rowb + mi * 16 + v);
          float hv = fmaxf(a[v] * scale + bias, 0.0f);
          outbuf[row * NTOT + col] = hv;
          lmin = fminf(lmin, hv); lmax = fmaxf(lmax, hv);
        }
      }
    }
    smin[tid] = lmin; smax[tid] = lmax; __syncthreads();
    for (int s = 128; s > 0; s >>= 1) {
      if (tid < s) { smin[tid] = fminf(smin[tid], smin[tid + s]); smax[tid] = fmaxf(smax[tid], smax[tid + s]); }
      __syncthreads();
    }
    if (tid == 0) { atomicMin(&scal[2], encf(smin[0])); atomicMax(&scal[3], encf(smax[0])); }
  } else {
#pragma unroll
    for (int mi = 0; mi < 2; ++mi) {
#pragma unroll
      for (int ni = 0; ni < 2; ++ni) {
        v8f a = (mi == 0) ? ((ni == 0) ? acc00 : acc01) : ((ni == 0) ? acc10 : acc11);
        const float bias = (ni == 0) ? bias0 : bias1;
        const int col = cb + ni * 16;
#pragma unroll
        for (int v = 0; v < 8; ++v) {
          const size_t row = (size_t)(rowb + mi * 16 + v);
          outbuf[row * NTOT + col] = a[v] * scale + bias;
        }
      }
    }
  }
}

// ---------------- gating + neighbor sum ----------------
__global__ void gate_kernel(const float* __restrict__ c2, const float* __restrict__ mask,
                            float* __restrict__ ns) {
  const int ba = blockIdx.x;      // 0..1535
  const int f  = threadIdx.x;     // 0..63
  float s = 0.0f;
  for (int n = 0; n < NBR; ++n) {
    const int m = ba * NBR + n;
    float g = c2[(size_t)m * N2 + f];
    float e = c2[(size_t)m * N2 + FN + f];
    float sg = 1.0f / (1.0f + expf(-g));
    s += sg * tanhf(e) * mask[m];
  }
  ns[(size_t)ba * FN + f] = s;
}

// ---------------- batchnorm stats + final output ----------------
__global__ void final_kernel(const float* __restrict__ node, const float* __restrict__ ns,
                             const float* __restrict__ gamma, const float* __restrict__ beta,
                             float* __restrict__ out) {
  __shared__ float ssum[256], ssq[256];
  const int f = blockIdx.x;       // feature 0..63
  const int t = threadIdx.x;
  float s = 0.0f, q = 0.0f;
  for (int r = t; r < NBA; r += 256) {
    float v = ns[(size_t)r * FN + f];
    s += v; q += v * v;
  }
  ssum[t] = s; ssq[t] = q; __syncthreads();
  for (int st = 128; st > 0; st >>= 1) {
    if (t < st) { ssum[t] += ssum[t + st]; ssq[t] += ssq[t + st]; }
    __syncthreads();
  }
  const float mean = ssum[0] * (1.0f / (float)NBA);
  const float var  = ssq[0] * (1.0f / (float)NBA) - mean * mean;   // biased
  const float inv  = rsqrtf(var + 1e-5f);
  const float g = gamma[f], bb = beta[f];
  for (int r = t; r < NBA; r += 256) {
    float v = (ns[(size_t)r * FN + f] - mean) * inv * g + bb;
    float o = node[(size_t)r * FN + f] + v;
    out[(size_t)r * FN + f] = fmaxf(o, 0.0f);
  }
}

// ---------------- launch ----------------
extern "C" void kernel_launch(void* const* d_in, const int* in_sizes, int n_in,
                              void* d_out, int out_size, void* d_ws, size_t ws_size,
                              hipStream_t stream) {
  const float* node  = (const float*)d_in[0];   // (16,96,64)
  const float* edge  = (const float*)d_in[1];   // (16,96,12,64)
  const float* mask  = (const float*)d_in[2];   // (16,96,12)
  const float* cond1 = (const float*)d_in[3];   // (2,512,128)
  const float* cond2 = (const float*)d_in[4];   // (2,128,512)
  const float* eps1  = (const float*)d_in[5];   // (8,512,128)
  const float* eps2  = (const float*)d_in[6];   // (8,128,512)
  const float* gamma = (const float*)d_in[7];   // (64,)
  const float* beta  = (const float*)d_in[8];   // (64,)
  float* out = (float*)d_out;

  char* ws = (char*)d_ws;
  unsigned*      scal = (unsigned*)(ws + O_SCAL);
  _Float16*      w1t  = (_Float16*)(ws + O_W1);
  _Float16*      w2t  = (_Float16*)(ws + O_W2);
  float*         rs1  = (float*)(ws + O_RS1);
  float*         rs2  = (float*)(ws + O_RS2);
  unsigned char* q1   = (unsigned char*)(ws + O_Q1);
  unsigned char* q2   = (unsigned char*)(ws + O_Q2);
  float*         hbuf = (float*)(ws + O_H);
  float*         c2b  = (float*)(ws + O_C2);
  float*         ns   = (float*)(ws + O_NS);

  init_kernel<<<1, 32, 0, stream>>>(scal);
  minmax_kernel<<<512, 256, 0, stream>>>(node, BATCH * AT * FN, edge, MROWS * FN, scal);

  prepw_kernel<<<(N1 * K1 + 255) / 256, 256, 0, stream>>>(cond1, eps1, w1t, N1, I1);
  prepw_kernel<<<(N2 * K2 + 255) / 256, 256, 0, stream>>>(cond2, eps2, w2t, N2, I2);
  rowsum_kernel<<<2, 256, 0, stream>>>(cond1, eps1 + (size_t)7 * N1 * I1, rs1, N1, I1, 1);
  rowsum_kernel<<<1, 128, 0, stream>>>(cond2 + (size_t)N2 * I2, nullptr, rs2, N2, I2, 0);

  quant1_kernel<<<(MROWS * I1 + 255) / 256, 256, 0, stream>>>(node, edge, q1, scal);
  gemm_kernel<I1, N1, true><<<dim3(MROWS / 64, N1 / 128), 256, 0, stream>>>(q1, w1t, rs1, hbuf, scal);
  quant2_kernel<<<(MROWS * I2 + 255) / 256, 256, 0, stream>>>(hbuf, q2, scal);
  gemm_kernel<I2, N2, false><<<dim3(MROWS / 64, N2 / 128), 256, 0, stream>>>(q2, w2t, rs2, c2b, scal);

  gate_kernel<<<NBA, 64, 0, stream>>>(c2b, mask, ns);
  final_kernel<<<FN, 256, 0, stream>>>(node, ns, gamma, beta, out);
}
